// Transformer_88510686036849
// MI455X (gfx1250) — compile-verified, matrix operand order checked by reading, not run on hardware
//
#include <hip/hip_runtime.h>
#include <hip/hip_bf16.h>

// ---------------------------------------------------------------------------
// Transformer forward (enc-dec) for MI455X (gfx1250, wave32, WMMA).
// All matmuls -> v_wmma_f32_16x16x32_bf16 (f32 -> bf16 in LDS, f32 accum).
// LDS tiles are K-contiguous for BOTH operands so each WMMA fragment is two
// ds_load_b128 whose bytes match the v16bf operand layout exactly.
// bf16 packing = 1x v_perm_b32 per pair (+ round bias), no half-reg moves.
// ---------------------------------------------------------------------------

#define cB     4
#define cS     512
#define cD     512
#define cH     8
#define cDFF   2048
#define cV     32000
#define cDEPTH 64
#define cMT    2048          // B*S token rows

#define BM 64
#define BN 128
#define BK 32
#define AP (BK + 16)         // LDS row stride: 48 elems = 96 B (16B aligned)

typedef __attribute__((ext_vector_type(16))) __bf16 v16bf;
typedef __attribute__((ext_vector_type(2)))  __bf16 v2bf;
typedef __attribute__((ext_vector_type(8)))  float  v8f;

// pack two floats -> two bf16 in one dword
__device__ __forceinline__ unsigned pkbf(float a, float b) {
#if __has_builtin(__builtin_amdgcn_cvt_pk_bf16_f32)
    v2bf r = __builtin_amdgcn_cvt_pk_bf16_f32(a, b);
    return __builtin_bit_cast(unsigned, r);
#else
    // round-half-up then grab the two high halves with one v_perm_b32
    unsigned ua = __builtin_bit_cast(unsigned, a) + 0x8000u;
    unsigned ub = __builtin_bit_cast(unsigned, b) + 0x8000u;
    return __builtin_amdgcn_perm(ub, ua, 0x07060302u);
#endif
}

// ---------------------------------------------------------------------------
// Batched GEMM: C[b,h] = A (MxK) * B (KxN) (+bias)(+relu).
// TRANSB==1: B stored (N x K) row-major (Q*K^T path).
// 256 threads = 8 waves; block tile 64x128; wave tile 32x32 = 2x2 WMMA.
// Requirements (satisfied here): M%64==0, K%32==0, ld%4==0, 16B base align.
// N may be < BN: OOB reads land in workspace, writes are guarded.
// ---------------------------------------------------------------------------
template<int TRANSB>
__global__ __launch_bounds__(256) void gemm_wmma(
    const float* __restrict__ A, long long lda, long long sAb, long long sAh,
    const float* __restrict__ Bm, long long ldb, long long sBb, long long sBh,
    float* __restrict__ C, long long ldc, long long sCb, long long sCh,
    const float* __restrict__ bias, int relu,
    int N, int Kd, int batH)
{
    __shared__ __bf16 As[BM][AP];                // K-contiguous rows
    __shared__ __bf16 Bs[BN][AP];                // K-contiguous rows (both modes)

    const int bz = blockIdx.z;
    const int bb = bz / batH;
    const int hh = bz % batH;
    A  += (long long)bb * sAb + (long long)hh * sAh;
    Bm += (long long)bb * sBb + (long long)hh * sBh;
    C  += (long long)bb * sCb + (long long)hh * sCh;

    const int tid    = threadIdx.x;
    const int lane   = tid & 31;
    const int wid    = tid >> 5;
    const int wm     = wid & 1;      // 2 waves along M  -> 64 rows
    const int wn     = wid >> 1;     // 4 waves along N  -> 128 cols
    const int m0     = blockIdx.y * BM;
    const int n0     = blockIdx.x * BN;
    const int laneHi = lane >> 4;
    const int lane16 = lane & 15;

    v8f zero = {};
    v8f acc[2][2];
    acc[0][0] = zero; acc[0][1] = zero; acc[1][0] = zero; acc[1][1] = zero;

    float4 aReg[2];                  // 64x32 tile: 8 floats/thread
    float4 bReg[4];                  // 128x32 tile: 16 floats/thread

    auto loadA = [&](int kk) {
        int r  = tid >> 2;                       // 0..63
        int c8 = (tid & 3) << 3;                 // 0,8,16,24
        const float* p = A + (long long)(m0 + r) * lda + kk + c8;
        aReg[0] = *(const float4*)(p);
        aReg[1] = *(const float4*)(p + 4);
    };
    auto loadB = [&](int kk) {
        #pragma unroll
        for (int i = 0; i < 2; ++i) {
            int g = tid + i * 256;               // 0..511
            if (TRANSB) {
                int n  = g >> 2;                 // 0..127
                int c8 = (g & 3) << 3;           // 0,8,16,24
                const float* p = Bm + (long long)(n0 + n) * ldb + kk + c8;
                bReg[2 * i + 0] = *(const float4*)(p);
                bReg[2 * i + 1] = *(const float4*)(p + 4);
            } else {
                int k = (g >> 5) << 1;           // even k: 0..30
                int n = (g & 31) << 2;           // 0..124
                const float* p = Bm + (long long)(kk + k) * ldb + n0 + n;
                bReg[2 * i + 0] = *(const float4*)(p);        // row k
                bReg[2 * i + 1] = *(const float4*)(p + ldb);  // row k+1
            }
        }
    };
    auto stage = [&]() {
        {   // A: one ds_store_b128 per thread
            int r  = tid >> 2;
            int c8 = (tid & 3) << 3;
            uint4 q;
            q.x = pkbf(aReg[0].x, aReg[0].y);
            q.y = pkbf(aReg[0].z, aReg[0].w);
            q.z = pkbf(aReg[1].x, aReg[1].y);
            q.w = pkbf(aReg[1].z, aReg[1].w);
            *(uint4*)&As[r][c8] = q;
        }
        #pragma unroll
        for (int i = 0; i < 2; ++i) {
            int g = tid + i * 256;
            if (TRANSB) {                        // ds_store_b128
                int n  = g >> 2;
                int c8 = (g & 3) << 3;
                uint4 q;
                q.x = pkbf(bReg[2 * i + 0].x, bReg[2 * i + 0].y);
                q.y = pkbf(bReg[2 * i + 0].z, bReg[2 * i + 0].w);
                q.z = pkbf(bReg[2 * i + 1].x, bReg[2 * i + 1].y);
                q.w = pkbf(bReg[2 * i + 1].z, bReg[2 * i + 1].w);
                *(uint4*)&Bs[n][c8] = q;
            } else {                             // transpose: 4x ds_store_b32
                int k = (g >> 5) << 1;
                int n = (g & 31) << 2;
                float4 r0 = bReg[2 * i + 0], r1 = bReg[2 * i + 1];
                *(unsigned*)&Bs[n + 0][k] = pkbf(r0.x, r1.x);
                *(unsigned*)&Bs[n + 1][k] = pkbf(r0.y, r1.y);
                *(unsigned*)&Bs[n + 2][k] = pkbf(r0.z, r1.z);
                *(unsigned*)&Bs[n + 3][k] = pkbf(r0.w, r1.w);
            }
        }
    };

    union V16 { uint4 q[2]; v16bf v; };

    loadA(0);
    loadB(0);
    for (int kk = 0; kk < Kd; kk += BK) {
        stage();
        __syncthreads();
        if (kk + BK < Kd) {                      // prefetch next tile
            loadA(kk + BK);
            loadB(kk + BK);
        }

        // Fragment = two ds_load_b128; byte order == v16bf operand layout:
        // e0..7  -> K = laneHi*8 .. +7 ; e8..15 -> K = 16+laneHi*8 .. +7
        v16bf afrag[2], bfrag[2];
        #pragma unroll
        for (int i = 0; i < 2; ++i) {
            int row = wm * 32 + i * 16 + lane16;
            V16 u;
            u.q[0] = *(const uint4*)&As[row][(laneHi << 3)];
            u.q[1] = *(const uint4*)&As[row][16 + (laneHi << 3)];
            afrag[i] = u.v;
        }
        #pragma unroll
        for (int j = 0; j < 2; ++j) {
            int col = wn * 32 + j * 16 + lane16;
            V16 u;
            u.q[0] = *(const uint4*)&Bs[col][(laneHi << 3)];
            u.q[1] = *(const uint4*)&Bs[col][16 + (laneHi << 3)];
            bfrag[j] = u.v;
        }

        #pragma unroll
        for (int i = 0; i < 2; ++i)
            #pragma unroll
            for (int j = 0; j < 2; ++j)
                acc[i][j] = __builtin_amdgcn_wmma_f32_16x16x32_bf16(
                    false, afrag[i], false, bfrag[j],
                    (short)0, acc[i][j], false, false);
        __syncthreads();
    }

    // Epilogue: lane -> N = lane&15, VGPR r -> M = r + 8*(lane>>4).
    // Stores from lanes 0..15 / 16..31 cover consecutive 64B row segments.
    #pragma unroll
    for (int i = 0; i < 2; ++i) {
        #pragma unroll
        for (int j = 0; j < 2; ++j) {
            int nc = n0 + wn * 32 + j * 16 + lane16;
            if (nc < N) {
                float bv = (bias != nullptr) ? bias[nc] : 0.f;
                #pragma unroll
                for (int r = 0; r < 8; ++r) {
                    int mr = m0 + wm * 32 + i * 16 + r + (laneHi << 3);
                    float v = acc[i][j][r] + bv;
                    if (relu) v = fmaxf(v, 0.f);
                    C[(long long)mr * ldc + nc] = v;
                }
            }
        }
    }
}

// ---------------------------------------------------------------------------
// Row softmax with fused 1/sqrt(depth) scale and masking.
// mode 0: none, 1: causal (k>q masked), 2: key-padding (target_seq[b,k]==0).
// grid = (S, B*H), block = 256, row length S = 512 -> 2 elems/thread.
// ---------------------------------------------------------------------------
__global__ __launch_bounds__(256) void softmax_mask(
    float* __restrict__ sc, const int* __restrict__ tgt, int mode)
{
    const int q  = blockIdx.x;
    const int bh = blockIdx.y;
    const int b  = bh >> 3;                       // H == 8
    float* row = sc + ((long long)bh * cS + q) * cS;
    const int tid = threadIdx.x;
    __shared__ float red[256];

    float v[2];
    #pragma unroll
    for (int i = 0; i < 2; ++i) {
        int k = tid + i * 256;
        float x = row[k] * 0.125f;                // 1/sqrt(64)
        if (mode == 1 && k > q) x = -1e9f;
        if (mode == 2 && tgt[b * cS + k] == 0) x = -1e9f;
        v[i] = x;
    }
    red[tid] = fmaxf(v[0], v[1]);
    __syncthreads();
    for (int s = 128; s > 0; s >>= 1) {
        if (tid < s) red[tid] = fmaxf(red[tid], red[tid + s]);
        __syncthreads();
    }
    float m = red[0];
    __syncthreads();
    float e0 = __expf(v[0] - m), e1 = __expf(v[1] - m);
    red[tid] = e0 + e1;
    __syncthreads();
    for (int s = 128; s > 0; s >>= 1) {
        if (tid < s) red[tid] += red[tid + s];
        __syncthreads();
    }
    float inv = 1.0f / red[0];
    row[tid]       = e0 * inv;
    row[tid + 256] = e1 * inv;
}

// ---------------------------------------------------------------------------
// out = LayerNorm(x + a) * g + b       (row length D = 512, 2 elems/thread)
// ---------------------------------------------------------------------------
__global__ __launch_bounds__(256) void add_layernorm(
    const float* __restrict__ x, const float* __restrict__ a,
    const float* __restrict__ g, const float* __restrict__ bt,
    float* __restrict__ out)
{
    const int row = blockIdx.x;
    const int tid = threadIdx.x;
    const float* xr = x + (long long)row * cD;
    const float* ar = a + (long long)row * cD;
    __shared__ float red[256];

    float v0 = xr[tid] + ar[tid];
    float v1 = xr[tid + 256] + ar[tid + 256];
    red[tid] = v0 + v1;
    __syncthreads();
    for (int s = 128; s > 0; s >>= 1) {
        if (tid < s) red[tid] += red[tid + s];
        __syncthreads();
    }
    float mean = red[0] * (1.0f / cD);
    __syncthreads();
    float d0 = v0 - mean, d1 = v1 - mean;
    red[tid] = d0 * d0 + d1 * d1;
    __syncthreads();
    for (int s = 128; s > 0; s >>= 1) {
        if (tid < s) red[tid] += red[tid + s];
        __syncthreads();
    }
    float rstd = rsqrtf(red[0] * (1.0f / cD) + 1e-5f);
    float* orow = out + (long long)row * cD;
    orow[tid]       = d0 * rstd * g[tid]       + bt[tid];
    orow[tid + 256] = d1 * rstd * g[tid + 256] + bt[tid + 256];
}

// ---------------------------------------------------------------------------
// out[row, d] = emb[ids[row], d] * sqrt(D) + PE(pos = row % S, d)
// ---------------------------------------------------------------------------
__global__ __launch_bounds__(256) void embed_pe(
    const int* __restrict__ ids, const float* __restrict__ emb,
    float* __restrict__ out)
{
    const int row = blockIdx.x;
    const int s   = row % cS;
    const int tok = ids[row];
    const float scale = 22.62741699796952f;        // sqrt(512)
    const int tid = threadIdx.x;
    for (int d = tid; d < cD; d += 256) {
        float ang = (float)s *
            __expf((float)(d & ~1) * (-9.210340371976184f / (float)cD));
        float pe = (d & 1) ? __cosf(ang) : __sinf(ang);
        out[(long long)row * cD + d] = emb[(long long)tok * cD + d] * scale + pe;
    }
}

// ---------------------------------------------------------------------------
// Host orchestration
// ---------------------------------------------------------------------------
extern "C" void kernel_launch(void* const* d_in, const int* in_sizes, int n_in,
                              void* d_out, int out_size, void* d_ws, size_t ws_size,
                              hipStream_t stream)
{
    auto F = [&](int i) -> const float* { return (const float*)d_in[i]; };
    const int*   inp_ids = (const int*)d_in[0];
    const int*   tgt_ids = (const int*)d_in[1];
    const float* emb     = F(2);

    // Input flattening (depth-first, dict order):
    //   enc layer l base = 3 + 16*l:
    //     wq bq wk bk wv bv wo bo | ln1.g ln1.b ln2.g ln2.b | ff1.w ff1.b ff2.w ff2.b
    //   dec layer l base = 67 + 26*l:
    //     attn1[8] attn2[8] ln1[2] ln2[2] ln3[2] ff1[2] ff2[2]
    //   171: w_out, 172: b_out

    // Workspace layout (floats); total 18 * 2^20 floats = 72 MB
    const long long MT = (long long)cMT * cD;       // 2^20
    float* ws = (float*)d_ws;
    float* X  = ws;                                  // enc activations / enc_out
    float* Y  = ws + 1 * MT;                         // dec activations
    float* Qb = ws + 2 * MT;                         // q / ctx
    float* Kb = ws + 3 * MT;
    float* Vb = ws + 4 * MT;
    float* Tb = ws + 5 * MT;                         // mha/ffn output tmp
    float* Hh = ws + 6 * MT;                         // FFN hidden (4*MT)
    float* SC = ws + 10 * MT;                        // encoder attn probs (8*MT)

    // Output layout: logits [B,S,V] | self_attns [LD,B,H,S,S] | encdec [LD,...]
    float* out      = (float*)d_out;
    float* logits   = out;
    const long long ATT = (long long)cB * cH * cS * cS;   // 8,388,608
    float* self0    = out + (long long)cMT * cV;          // 65,536,000
    float* cross0   = self0 + 4 * ATT;

    // ---- GEMM wrappers ----
    auto proj = [&](const float* inp, int lda, const float* w, const float* bias,
                    float* outp, int N, int K, int relu) {
        dim3 g((N + BN - 1) / BN, cMT / BM, 1);
        gemm_wmma<0><<<g, 256, 0, stream>>>(
            inp, lda, 0, 0, w, N, 0, 0,
            outp, N, 0, 0, bias, relu, N, K, 1);
    };
    auto scores_gemm = [&](const float* q, const float* k, float* sc) {
        dim3 g(cS / BN, cS / BM, cB * cH);
        gemm_wmma<1><<<g, 256, 0, stream>>>(
            q, cD, (long long)cS * cD, cDEPTH,
            k, cD, (long long)cS * cD, cDEPTH,
            sc, cS, (long long)cH * cS * cS, (long long)cS * cS,
            nullptr, 0, cS, cDEPTH, cH);
    };
    auto ctx_gemm = [&](const float* sc, const float* v, float* ctx) {
        dim3 g(1, cS / BM, cB * cH);
        gemm_wmma<0><<<g, 256, 0, stream>>>(
            sc, cS, (long long)cH * cS * cS, (long long)cS * cS,
            v, cD, (long long)cS * cD, cDEPTH,
            ctx, cD, (long long)cS * cD, cDEPTH,
            nullptr, 0, cDEPTH, cS, cH);
    };
    auto smax = [&](float* sc, int mode) {
        dim3 g(cS, cB * cH);
        softmax_mask<<<g, 256, 0, stream>>>(sc, tgt_ids, mode);
    };
    auto aln = [&](const float* x, const float* a, const float* g_,
                   const float* b_, float* o) {
        add_layernorm<<<cMT, 256, 0, stream>>>(x, a, g_, b_, o);
    };

    // Full multi-head attention block with fused residual+LN.
    auto mha = [&](float* q_src, const float* kv_src, int pbase,
                   float* sc, int maskmode,
                   const float* lng, const float* lnb) {
        proj(q_src,  cD, F(pbase + 0), F(pbase + 1), Qb, cD, cD, 0);
        proj(kv_src, cD, F(pbase + 2), F(pbase + 3), Kb, cD, cD, 0);
        proj(kv_src, cD, F(pbase + 4), F(pbase + 5), Vb, cD, cD, 0);
        scores_gemm(Qb, Kb, sc);
        smax(sc, maskmode);
        ctx_gemm(sc, Vb, Qb);                       // reuse Qb as ctx
        proj(Qb, cD, F(pbase + 6), F(pbase + 7), Tb, cD, cD, 0);
        aln(q_src, Tb, lng, lnb, q_src);
    };
    auto ffn = [&](float* io, int wbase, const float* lng, const float* lnb) {
        proj(io, cD,   F(wbase + 0), F(wbase + 1), Hh, cDFF, cD, 1);   // relu
        proj(Hh, cDFF, F(wbase + 2), F(wbase + 3), Tb, cD, cDFF, 0);
        aln(io, Tb, lng, lnb, io);
    };

    // ================= Encoder =================
    embed_pe<<<cMT, 256, 0, stream>>>(inp_ids, emb, X);
    for (int l = 0; l < 4; ++l) {
        int base = 3 + 16 * l;
        mha(X, X, base, SC, /*mask*/0, F(base + 8), F(base + 9));      // ln1
        ffn(X, base + 12, F(base + 10), F(base + 11));                  // ln2
    }

    // ================= Decoder =================
    embed_pe<<<cMT, 256, 0, stream>>>(tgt_ids, emb, Y);
    for (int l = 0; l < 4; ++l) {
        int base = 67 + 26 * l;
        // self-attn (causal); probs are an output -> write into d_out slice
        mha(Y, Y, base + 0, self0 + (long long)l * ATT, /*mask*/1,
            F(base + 16), F(base + 17));                                // ln1
        // cross-attn (key padding mask); probs into d_out slice
        mha(Y, X, base + 8, cross0 + (long long)l * ATT, /*mask*/2,
            F(base + 18), F(base + 19));                                // ln2
        ffn(Y, base + 22, F(base + 20), F(base + 21));                  // ln3
    }

    // ================= Output projection =================
    proj(Y, cD, F(171), F(172), logits, cV, cD, 0);

    (void)in_sizes; (void)n_in; (void)out_size; (void)ws_size;
}
